// GraphLaplacianAttentionTransformer_87471303950498
// MI455X (gfx1250) — matrix operand order repaired and checked by reading, not compiled, not run on hardware
//
#include <hip/hip_runtime.h>
#include <cstdint>
#include <cstddef>

#define DEV __device__ __forceinline__

typedef __bf16 bf16;
typedef __attribute__((ext_vector_type(8)))  bf16  v8bf;
typedef __attribute__((ext_vector_type(16))) bf16  v16bf;
typedef __attribute__((ext_vector_type(8)))  float v8f;

constexpr int N_TOK = 8192;
constexpr int DIM   = 2048;
constexpr int HD    = 512;
constexpr float LAM = 0.1f;
// sigma = 0.5 -> W = exp(-d2 / (2*sigma^2)) = exp(-2*d2)
constexpr float GAUSS_COEF = 2.0f;
constexpr float ATTN_SCALE = 0.04419417382415922f;  // 1/sqrt(512)

// ---------------- CDNA5 async global->LDS staging ----------------
DEV void async_ld_b128(const bf16* g, bf16* l) {
  // VDST = VGPR holding LDS byte address, VADDR = 64-bit global address.
  unsigned loff = (unsigned)(unsigned long long)l;  // low 32 bits of generic ptr = LDS offset
  asm volatile("global_load_async_to_lds_b128 %0, %1, off"
               :: "v"(loff), "v"(g) : "memory");
}
DEV void wait_async0() {
  asm volatile("s_wait_asynccnt 0" ::: "memory");
}
DEV v16bf cat8(v8bf lo, v8bf hi) {
  return __builtin_shufflevector(lo, hi, 0,1,2,3,4,5,6,7,8,9,10,11,12,13,14,15);
}

// ---------------- epilogues (32-bit offset math; all offsets < 2^31) ----------------
struct EpiGram {            // acc = x_i . x_j  ->  W = exp(-2*max(d2,0))
  const float* sq; float* W; unsigned n;
  DEV void emit(float g, unsigned gm, unsigned gn) const {
    float d2 = fmaxf(sq[gm] + sq[gn] - 2.0f * g, 0.0f);
    W[gm * n + gn] = __expf(-GAUSS_COEF * d2);
  }
};
struct EpiQKV {             // acc + bias -> bf16, row-major
  const float* bias; bf16* out; unsigned ld;
  DEV void emit(float a, unsigned gm, unsigned gn) const {
    out[gm * ld + gn] = (bf16)(a + bias[gn]);
  }
};
struct EpiVT {              // acc + bias -> bf16, transposed store (V' -> [H][N])
  const float* bias; bf16* out; unsigned ld;
  DEV void emit(float a, unsigned gm, unsigned gn) const {
    out[gn * ld + gm] = (bf16)(a + bias[gn]);
  }
};
struct EpiQK {              // logits = scale*acc - lam*W (+ lam*rowsumW on diag), in place over W
  float* S; const float* rsW; float scale, lam; unsigned n;
  DEV void emit(float a, unsigned gm, unsigned gn) const {
    unsigned idx = gm * n + gn;
    float l = scale * a - lam * S[idx];
    l += (gm == gn) ? lam * rsW[gm] : 0.0f;
    S[idx] = l;
  }
};
struct EpiPV {              // normalize by softmax denominator, store bf16
  const float* rse; bf16* Hb; unsigned ld;
  DEV void emit(float a, unsigned gm, unsigned gn) const {
    Hb[gm * ld + gn] = (bf16)(a / rse[gm]);
  }
};
struct EpiOut {             // + bo, f32 output
  const float* bo; float* out; unsigned ld;
  DEV void emit(float a, unsigned gm, unsigned gn) const {
    out[gm * ld + gn] = a + bo[gn];
  }
};

// ---------------- generic tiled GEMM:  C[M x Nc] = A[M x K] * Bsrc[Nc x K]^T ----------------
// Block 128x128, 8 waves (256 thr) in a 4x2 wave grid, wave tile 32x64:
// per K-step each wave does 8 v_wmma_f32_16x16x32_bf16 fed by 12 ds_load_b128 (1.5 loads/WMMA).
// Double-buffered LDS staging via global_load_async_to_lds_b128 (ASYNCcnt).
template <class Epi>
__global__ __launch_bounds__(256) void gemm_abt(
    const bf16* __restrict__ A, const bf16* __restrict__ B, int K, Epi epi)
{
  constexpr int LD = 40;  // padded row stride (elements): 80B rows, 16B-aligned sub-chunks
  __shared__ __align__(16) bf16 lA[2][128 * LD];
  __shared__ __align__(16) bf16 lB[2][128 * LD];

  const unsigned gm0 = blockIdx.y * 128;
  const unsigned gn0 = blockIdx.x * 128;
  const int t    = threadIdx.x;
  const int lane = t & 31;
  const int wave = t >> 5;           // 0..7
  const int wm = (wave >> 1) * 32;   // 0,32,64,96
  const int wn = (wave & 1) * 64;    // 0,64
  const int r    = lane & 15;
  const int half = lane >> 4;

  // stage copy: thread t moves 32B of one A row and 32B of one B row per stage (4 async b128)
  const int crow  = t >> 1;          // 0..127
  const int chalf = t & 1;           // element offset 0 / 16
  const bf16* gA = A + (size_t)(gm0 + crow) * K + chalf * 16;
  const bf16* gB = B + (size_t)(gn0 + crow) * K + chalf * 16;
  const int loffs = crow * LD + chalf * 16;

  v8f c[2][4] = {};

  auto issue = [&](int buf, int k0) {
    const bf16* a = gA + k0;
    const bf16* b = gB + k0;
    bf16* la = &lA[buf][loffs];
    bf16* lb = &lB[buf][loffs];
    async_ld_b128(a,     la);
    async_ld_b128(a + 8, la + 8);
    async_ld_b128(b,     lb);
    async_ld_b128(b + 8, lb + 8);
  };

  const int nk = K >> 5;             // K is a multiple of 32 for every pass here
  issue(0, 0);
  wait_async0();
  __syncthreads();

  for (int kt = 0; kt < nk; ++kt) {
    const int cur = kt & 1;
    if (kt + 1 < nk) issue(cur ^ 1, (kt + 1) << 5);

    const bf16* bA = &lA[cur][0];
    const bf16* bB = &lB[cur][0];
    v16bf af[2], bfr[4];
#pragma unroll
    for (int mi = 0; mi < 2; ++mi) {
      // ISA 16-bit A 16x32 layout: lane(r,half) holds K = half*8+[0..7] and 16+half*8+[0..7]
      const bf16* rp = bA + (wm + mi * 16 + r) * LD;
      af[mi] = cat8(*(const v8bf*)(rp + half * 8),
                    *(const v8bf*)(rp + 16 + half * 8));
    }
#pragma unroll
    for (int ni = 0; ni < 4; ++ni) {
      // ISA 16-bit B 32x16 layout: lane(n,half) holds K = half*8+[0..7] and 16+half*8+[0..7]
      const bf16* rp = bB + (wn + ni * 16 + r) * LD;
      bfr[ni] = cat8(*(const v8bf*)(rp + half * 8),
                     *(const v8bf*)(rp + 16 + half * 8));
    }
#pragma unroll
    for (int mi = 0; mi < 2; ++mi)
#pragma unroll
      for (int ni = 0; ni < 4; ++ni)
        c[mi][ni] = __builtin_amdgcn_wmma_f32_16x16x32_bf16(
            false, af[mi], false, bfr[ni], (short)0, c[mi][ni], false, false);

    if (kt + 1 < nk) { wait_async0(); __syncthreads(); }
  }

  // C layout: element (m = rr + 8*half, n = lane&15) in VGPR rr
#pragma unroll
  for (int mi = 0; mi < 2; ++mi)
#pragma unroll
    for (int ni = 0; ni < 4; ++ni) {
      const unsigned gn = gn0 + wn + ni * 16 + r;
#pragma unroll
      for (int rr = 0; rr < 8; ++rr) {
        const unsigned gm = gm0 + wm + mi * 16 + rr + 8 * half;
        epi.emit(c[mi][ni][rr], gm, gn);
      }
    }
}

// ---------------- small prep / reduction kernels ----------------
__global__ __launch_bounds__(256) void prep_x(const float* __restrict__ x,
                                              bf16* __restrict__ xh,
                                              float* __restrict__ sq) {
  __shared__ float red[256];
  const int row = blockIdx.x, t = threadIdx.x;
  float acc = 0.0f;
  const float* xp = x + (size_t)row * DIM;
  bf16* hp = xh + (size_t)row * DIM;
  for (int d = t; d < DIM; d += 256) {
    float v = xp[d];
    hp[d] = (bf16)v;
    acc += v * v;
  }
  red[t] = acc; __syncthreads();
  for (int s = 128; s > 0; s >>= 1) {
    if (t < s) red[t] += red[t + s];
    __syncthreads();
  }
  if (t == 0) sq[row] = red[0];
}

__global__ void cvt_bf16(const float* __restrict__ s, bf16* __restrict__ d, int n) {
  int i = blockIdx.x * 256 + threadIdx.x;
  if (i < n) d[i] = (bf16)s[i];
}

// Weff = L_theta @ Wsrc   (H x D), output bf16
__global__ __launch_bounds__(256) void fold_theta(
    const float* __restrict__ L,
    const float* __restrict__ w0, const float* __restrict__ w1, const float* __restrict__ w2,
    bf16* __restrict__ o0, bf16* __restrict__ o1, bf16* __restrict__ o2)
{
  __shared__ float sL[16][HD];
  const int z = blockIdx.z;
  const float* Ws = (z == 0) ? w0 : (z == 1) ? w1 : w2;
  bf16* O = (z == 0) ? o0 : (z == 1) ? o1 : o2;
  const int h0 = blockIdx.y * 16;
  const int t  = threadIdx.x;
  for (int i = t; i < 16 * HD; i += 256)
    sL[i / HD][i % HD] = L[(size_t)(h0 + i / HD) * HD + (i % HD)];
  __syncthreads();
  const int d = blockIdx.x * 256 + t;
  float acc[16];
#pragma unroll
  for (int hh = 0; hh < 16; ++hh) acc[hh] = 0.0f;
  for (int j = 0; j < HD; ++j) {
    const float wv = Ws[(size_t)j * DIM + d];
#pragma unroll
    for (int hh = 0; hh < 16; ++hh) acc[hh] += sL[hh][j] * wv;
  }
#pragma unroll
  for (int hh = 0; hh < 16; ++hh)
    O[(size_t)(h0 + hh) * DIM + d] = (bf16)acc[hh];
}

__global__ void fold_bias(const float* __restrict__ L,
                          const float* b0, const float* b1, const float* b2,
                          float* o0, float* o1, float* o2) {
  const int z = blockIdx.x;
  const float* b = (z == 0) ? b0 : (z == 1) ? b1 : b2;
  float* o = (z == 0) ? o0 : (z == 1) ? o1 : o2;
  const int h = threadIdx.x;
  float acc = 0.0f;
  for (int j = 0; j < HD; ++j) acc += L[(size_t)h * HD + j] * b[j];
  o[h] = acc;
}

template <int MODE>  // 0 = sum, 1 = max
__global__ __launch_bounds__(256) void row_reduce(const float* __restrict__ M,
                                                  float* __restrict__ out, int n) {
  __shared__ float red[256];
  const int row = blockIdx.x, t = threadIdx.x;
  float acc = MODE ? -3.4e38f : 0.0f;
  const float* p = M + (size_t)row * n;
  for (int j = t; j < n; j += 256) {
    float v = p[j];
    acc = MODE ? fmaxf(acc, v) : (acc + v);
  }
  red[t] = acc; __syncthreads();
  for (int s = 128; s > 0; s >>= 1) {
    if (t < s) red[t] = MODE ? fmaxf(red[t], red[t + s]) : (red[t] + red[t + s]);
    __syncthreads();
  }
  if (t == 0) out[row] = red[0];
}

__global__ __launch_bounds__(256) void softmax_row(const float* __restrict__ S,
                                                   const float* __restrict__ rmax,
                                                   float* __restrict__ rse,
                                                   bf16* __restrict__ P, int n) {
  __shared__ float red[256];
  const int row = blockIdx.x, t = threadIdx.x;
  const float m = rmax[row];
  float acc = 0.0f;
  const float* sp = S + (size_t)row * n;
  bf16* pp = P + (size_t)row * n;
  for (int j = t; j < n; j += 256) {
    float e = __expf(sp[j] - m);
    pp[j] = (bf16)e;
    acc += e;
  }
  red[t] = acc; __syncthreads();
  for (int s = 128; s > 0; s >>= 1) {
    if (t < s) red[t] += red[t + s];
    __syncthreads();
  }
  if (t == 0) rse[row] = red[0];
}

// ---------------- launcher ----------------
extern "C" void kernel_launch(void* const* d_in, const int* in_sizes, int n_in,
                              void* d_out, int out_size, void* d_ws, size_t ws_size,
                              hipStream_t stream) {
  const float* x  = (const float*)d_in[0];
  const float* wq = (const float*)d_in[1];
  const float* bq = (const float*)d_in[2];
  const float* wk = (const float*)d_in[3];
  const float* bk = (const float*)d_in[4];
  const float* wv = (const float*)d_in[5];
  const float* bv = (const float*)d_in[6];
  const float* lt = (const float*)d_in[7];
  const float* wo = (const float*)d_in[8];
  const float* bo = (const float*)d_in[9];
  float* out = (float*)d_out;

  char* p = (char*)d_ws;
  auto take = [&](size_t bytes) { char* r = p; p += (bytes + 255) & ~(size_t)255; return (void*)r; };
  bf16*  xh   = (bf16*) take((size_t)N_TOK * DIM * 2);
  float* sq   = (float*)take((size_t)N_TOK * 4);
  bf16*  wqe  = (bf16*) take((size_t)HD * DIM * 2);
  bf16*  wke  = (bf16*) take((size_t)HD * DIM * 2);
  bf16*  wve  = (bf16*) take((size_t)HD * DIM * 2);
  float* bqe  = (float*)take((size_t)HD * 4);
  float* bke  = (float*)take((size_t)HD * 4);
  float* bve  = (float*)take((size_t)HD * 4);
  bf16*  woh  = (bf16*) take((size_t)DIM * HD * 2);
  bf16*  qp   = (bf16*) take((size_t)N_TOK * HD * 2);
  bf16*  kp   = (bf16*) take((size_t)N_TOK * HD * 2);
  bf16*  vpT  = (bf16*) take((size_t)HD * N_TOK * 2);    // transposed V' for B-side of P@V
  float* S    = (float*)take((size_t)N_TOK * N_TOK * 4); // W, then logits (in place)
  bf16*  Pb   = (bf16*) take((size_t)N_TOK * N_TOK * 2); // probs (unnormalized exp)
  float* rsW  = (float*)take((size_t)N_TOK * 4);
  float* rmax = (float*)take((size_t)N_TOK * 4);
  float* rse  = (float*)take((size_t)N_TOK * 4);
  bf16*  Hb   = (bf16*) take((size_t)N_TOK * HD * 2);

  // --- prep ---
  prep_x<<<N_TOK, 256, 0, stream>>>(x, xh, sq);
  cvt_bf16<<<(DIM * HD + 255) / 256, 256, 0, stream>>>(wo, woh, DIM * HD);
  fold_theta<<<dim3(DIM / 256, HD / 16, 3), 256, 0, stream>>>(lt, wq, wk, wv, wqe, wke, wve);
  fold_bias<<<3, HD, 0, stream>>>(lt, bq, bk, bv, bqe, bke, bve);

  const dim3 blk(256);
  // --- Q'/K'/V' projections (bf16 WMMA, K = 2048) ---
  gemm_abt<EpiQKV><<<dim3(HD / 128, N_TOK / 128), blk, 0, stream>>>(xh, wqe, DIM, EpiQKV{bqe, qp, HD});
  gemm_abt<EpiQKV><<<dim3(HD / 128, N_TOK / 128), blk, 0, stream>>>(xh, wke, DIM, EpiQKV{bke, kp, HD});
  gemm_abt<EpiVT ><<<dim3(HD / 128, N_TOK / 128), blk, 0, stream>>>(xh, wve, DIM, EpiVT{bve, vpT, N_TOK});
  // --- Gram -> Gaussian affinity W (K = 2048, dominant 275 GF) ---
  gemm_abt<EpiGram><<<dim3(N_TOK / 128, N_TOK / 128), blk, 0, stream>>>(xh, xh, DIM, EpiGram{sq, S, N_TOK});
  row_reduce<0><<<N_TOK, 256, 0, stream>>>(S, rsW, N_TOK);        // degree = rowsum(W)
  // --- logits = scale*Q'K'^T - lam*W + lam*diag(rowsum) (K = 512, in place over W) ---
  gemm_abt<EpiQK><<<dim3(N_TOK / 128, N_TOK / 128), blk, 0, stream>>>(qp, kp, HD, EpiQK{S, rsW, ATTN_SCALE, LAM, N_TOK});
  // --- softmax (two row passes) ---
  row_reduce<1><<<N_TOK, 256, 0, stream>>>(S, rmax, N_TOK);
  softmax_row<<<N_TOK, 256, 0, stream>>>(S, rmax, rse, Pb, N_TOK);
  // --- H = (P @ V') / rowsumexp  (K = 8192) ---
  gemm_abt<EpiPV><<<dim3(HD / 128, N_TOK / 128), blk, 0, stream>>>(Pb, vpT, N_TOK, EpiPV{rse, Hb, HD});
  // --- out = H @ wo^T + bo  (K = 512, f32 out) ---
  gemm_abt<EpiOut><<<dim3(DIM / 128, N_TOK / 128), blk, 0, stream>>>(Hb, woh, HD, EpiOut{bo, out, DIM});
}
